// DocREModel_72662256714011
// MI455X (gfx1250) — matrix operand order repaired and checked by reading, not compiled, not run on hardware
//
#include <hip/hip_runtime.h>

// ---------------------------------------------------------------------------
// Problem constants (from the reference)
// ---------------------------------------------------------------------------
constexpr int B_   = 4;
constexpr int L_   = 1024;
constexpr int H_   = 768;
constexpr int NH_  = 12;
constexpr int E_   = 24;
constexpr int M_   = 4;
constexpr int P_   = 552;
constexpr int NL_  = 97;
constexpr int NLP_ = 112;             // NL padded to a multiple of 16
constexpr int BLK_ = 64;
constexpr int NBLK = H_ / BLK_;       // 12 bilinear blocks
constexpr int K2H_ = 2 * H_;          // 1536, K of the zh/zt GEMM
constexpr int KTOT = H_ * BLK_;       // 49152, K of the bilinear GEMM

typedef __attribute__((ext_vector_type(16))) _Float16 v16h;
typedef __attribute__((ext_vector_type(8)))  _Float16 v8h;
typedef __attribute__((ext_vector_type(8)))  float    v8f;

// ---------------------------------------------------------------------------
// Fragment builders: every fragment is two b128 loads into adjacent VGPRs.
// 16-bit A 16x32 layout: lane<16 -> K {0..7,16..23}; lane>=16 -> +8 (caller
// folds the +8 into the pointer). B 32x16: same K layout as A.
// ---------------------------------------------------------------------------
__device__ __forceinline__ v16h fragA(const _Float16* __restrict__ p) {
  v8h lo = *reinterpret_cast<const v8h*>(p);
  v8h hi = *reinterpret_cast<const v8h*>(p + 16);
  return __builtin_shufflevector(lo, hi, 0, 1, 2, 3, 4, 5, 6, 7,
                                 8, 9, 10, 11, 12, 13, 14, 15);
}
__device__ __forceinline__ v16h fragB(const _Float16* __restrict__ p) {
  v8h lo = *reinterpret_cast<const v8h*>(p);
  v8h hi = *reinterpret_cast<const v8h*>(p + 16);
  return __builtin_shufflevector(lo, hi, 0, 1, 2, 3, 4, 5, 6, 7,
                                 8, 9, 10, 11, 12, 13, 14, 15);
}
__device__ __forceinline__ v8f wmma_f16(v16h a, v16h b, v8f c) {
  return __builtin_amdgcn_wmma_f32_16x16x32_f16(false, a, false, b,
                                                (short)0, c, false, false);
}

// ---------------------------------------------------------------------------
// Stage 1: entity embeddings (masked logsumexp, stored f16) + mention-averaged
// attention. grid = B*E blocks, 256 threads.
// ---------------------------------------------------------------------------
__global__ void entity_kernel(const float* __restrict__ seq,     // (B,L,H)
                              const float* __restrict__ att,     // (B,NH,L,L)
                              const float* __restrict__ mmask,   // (B,E,M)
                              const int*   __restrict__ midx,    // (B,E,M)
                              _Float16* __restrict__ eemb16,     // (B,E,H) f16
                              float* __restrict__ eatt) {        // (B,E,NH,L)
  const int be = blockIdx.x;
  const int b = be / E_, e = be % E_;
  __shared__ float sm[M_];
  __shared__ int   si[M_];
  __shared__ float sden;
  if (threadIdx.x < M_) {
    sm[threadIdx.x] = mmask[(b * E_ + e) * M_ + threadIdx.x];
    si[threadIdx.x] = midx[(b * E_ + e) * M_ + threadIdx.x];
  }
  __syncthreads();
  if (threadIdx.x == 0) {
    float d = 0.f;
    for (int m = 0; m < M_; ++m) d += sm[m];
    sden = d;
  }
  __syncthreads();
  const float inv_den = 1.f / sden;

  for (int h = threadIdx.x; h < H_; h += blockDim.x) {
    float mx = -3.0e38f;
    for (int m = 0; m < M_; ++m)
      if (sm[m] > 0.f)
        mx = fmaxf(mx, seq[((long)b * L_ + si[m]) * H_ + h]);
    float s = 0.f;
    for (int m = 0; m < M_; ++m)
      if (sm[m] > 0.f)
        s += expf(seq[((long)b * L_ + si[m]) * H_ + h] - mx);
    eemb16[(b * E_ + e) * H_ + h] = (_Float16)(mx + logf(s));
  }

  for (int t = threadIdx.x; t < NH_ * L_; t += blockDim.x) {
    const int nh = t / L_, l = t % L_;
    float s = 0.f;
    for (int m = 0; m < M_; ++m)
      if (sm[m] > 0.f)
        s += att[(((long)b * NH_ + nh) * L_ + si[m]) * L_ + l];
    eatt[((long)(b * E_ + e) * NH_ + nh) * L_ + l] = s * inv_den;
  }
}

// ---------------------------------------------------------------------------
// Stage 2: ht_att normalized over L, stored f16. grid = B*P, 256 threads.
// ---------------------------------------------------------------------------
__global__ void htatt_kernel(const float* __restrict__ eatt,   // (B,E,NH,L)
                             const int*   __restrict__ hts,    // (B,P,2)
                             _Float16* __restrict__ ht16) {    // (B,P,L) f16
  const int bp = blockIdx.x;
  const int b = bp / P_;
  const int h0 = hts[bp * 2 + 0], h1 = hts[bp * 2 + 1];
  const float* A  = eatt + (long)(b * E_ + h0) * NH_ * L_;
  const float* Bm = eatt + (long)(b * E_ + h1) * NH_ * L_;
  _Float16* row = ht16 + (long)bp * L_;

  float vals[L_ / 256];
  float local = 0.f;
#pragma unroll
  for (int it = 0; it < L_ / 256; ++it) {
    const int l = threadIdx.x + it * 256;
    float s = 0.f;
#pragma unroll
    for (int nh = 0; nh < NH_; ++nh) s += A[nh * L_ + l] * Bm[nh * L_ + l];
    const float v = s * (1.f / NH_);
    vals[it] = v;
    local += v;
  }
  __shared__ float red[256];
  red[threadIdx.x] = local;
  __syncthreads();
  for (int st = 128; st > 0; st >>= 1) {
    if (threadIdx.x < st) red[threadIdx.x] += red[threadIdx.x + st];
    __syncthreads();
  }
  const float scale = 1.f / (red[0] + 1e-5f);
#pragma unroll
  for (int it = 0; it < L_ / 256; ++it)
    row[threadIdx.x + it * 256] = (_Float16)(vals[it] * scale);
}

// ---------------------------------------------------------------------------
// seq (B,L,H) f32 -> seqT (B,H,L) f16, LDS-tiled transpose.
// grid = (L/32, H/32, B), block = (32,8)
// ---------------------------------------------------------------------------
__global__ void transpose_cvt_kernel(const float* __restrict__ seq,
                                     _Float16* __restrict__ seqT) {
  __shared__ float tile[32][33];
  const int l0 = blockIdx.x * 32, h0 = blockIdx.y * 32, b = blockIdx.z;
  const int tx = threadIdx.x, ty = threadIdx.y;
#pragma unroll
  for (int j = 0; j < 4; ++j)
    tile[ty + j * 8][tx] = seq[((long)b * L_ + l0 + ty + j * 8) * H_ + h0 + tx];
  __syncthreads();
#pragma unroll
  for (int j = 0; j < 4; ++j)
    seqT[((long)b * H_ + h0 + ty + j * 8) * L_ + l0 + tx] =
        (_Float16)tile[tx][ty + j * 8];
}

// ---------------------------------------------------------------------------
// f32 -> f16 conversions
// ---------------------------------------------------------------------------
__global__ void cvt_f16_kernel(const float* __restrict__ in,
                               _Float16* __restrict__ out, int n) {
  const int i = blockIdx.x * blockDim.x + threadIdx.x;
  if (i < n) out[i] = (_Float16)in[i];
}
// Wb (97,KTOT) f32 -> (112,KTOT) f16, zero-padded rows.
__global__ void cvt_wb_kernel(const float* __restrict__ in,
                              _Float16* __restrict__ out) {
  const long i = (long)blockIdx.x * blockDim.x + threadIdx.x;
  if (i >= (long)NLP_ * KTOT) return;
  const int row = (int)(i / KTOT);
  out[i] = (row < NL_) ? (_Float16)in[i] : (_Float16)0.f;
}

// ---------------------------------------------------------------------------
// Stage 3: rs = ht @ seq  (P x L)x(L x H), 2x M-blocked: one wave computes a
// 32x16 C tile; the B fragment (seqT) is shared by both row subtiles.
// grid = (H/16, ceil(P/32), B), block = 32
// ---------------------------------------------------------------------------
__global__ void rs_gemm_kernel(const _Float16* __restrict__ ht16,  // (B,P,L)
                               const _Float16* __restrict__ seqT,  // (B,H,L)
                               _Float16* __restrict__ rs16) {      // (B,P,H)
  const int tn = blockIdx.x, tm = blockIdx.y, b = blockIdx.z;
  const int lane = threadIdx.x;
  const int mrow = lane & 15, hi = lane >> 4;
  const int kbA = hi ? 8 : 0, kbB = hi ? 8 : 0;
  const int col = tn * 16 + mrow;
  const int row0 = tm * 32 + mrow, row1 = row0 + 16;
  const int r0c = row0 < P_ ? row0 : 0;   // clamped rows only feed guarded C rows
  const int r1c = row1 < P_ ? row1 : 0;
  const _Float16* a0 = ht16 + ((long)b * P_ + r0c) * L_ + kbA;
  const _Float16* a1 = ht16 + ((long)b * P_ + r1c) * L_ + kbA;
  const _Float16* br = seqT + ((long)b * H_ + col) * L_ + kbB;

  v8f acc0 = {}, acc1 = {};
  for (int k0 = 0; k0 < L_; k0 += 32) {
    const v16h bm = fragB(br + k0);
    acc0 = wmma_f16(fragA(a0 + k0), bm, acc0);
    acc1 = wmma_f16(fragA(a1 + k0), bm, acc1);
  }
#pragma unroll
  for (int r = 0; r < 8; ++r) {
    const int o0 = tm * 32 + hi * 8 + r;
    if (o0 < P_)      rs16[((long)b * P_ + o0) * H_ + col]      = (_Float16)acc0[r];
    if (o0 + 16 < P_) rs16[((long)b * P_ + o0 + 16) * H_ + col] = (_Float16)acc1[r];
  }
}

// ---------------------------------------------------------------------------
// Stage 4: z = tanh(concat(hs, rs) @ W.T + bias), 2x M-blocked (W fragment
// shared), K loop split at H (hs vs rs source is loop-uniform).
// M = 2208 = 69 tiles of 32. grid = (48, 69), block = 32
// ---------------------------------------------------------------------------
__global__ void z_gemm_kernel(const _Float16* __restrict__ eemb16, // (B,E,H)
                              const _Float16* __restrict__ rs16,   // (B,P,H)
                              const _Float16* __restrict__ W16,    // (H,2H)
                              const float* __restrict__ bias,      // (H,)
                              const int*   __restrict__ hts,       // (B,P,2)
                              int sel,
                              _Float16* __restrict__ zout) {       // (B*P,H)
  const int tn = blockIdx.x, tm = blockIdx.y;
  const int lane = threadIdx.x;
  const int mrow = lane & 15, hi = lane >> 4;
  const int kbA = hi ? 8 : 0, kbB = hi ? 8 : 0;
  const int g0 = tm * 32 + mrow, g1 = g0 + 16;     // both < 2208
  const int b0 = g0 / P_, p0 = g0 % P_;
  const int b1 = g1 / P_, p1 = g1 % P_;
  const int e0 = hts[(b0 * P_ + p0) * 2 + sel];
  const int e1 = hts[(b1 * P_ + p1) * 2 + sel];
  const _Float16* hs0 = eemb16 + (long)(b0 * E_ + e0) * H_ + kbA;
  const _Float16* hs1 = eemb16 + (long)(b1 * E_ + e1) * H_ + kbA;
  const _Float16* rs0 = rs16 + (long)g0 * H_ + kbA;
  const _Float16* rs1 = rs16 + (long)g1 * H_ + kbA;
  const int col = tn * 16 + mrow;
  const _Float16* wr = W16 + (long)col * K2H_ + kbB;

  v8f acc0 = {}, acc1 = {};
  for (int k0 = 0; k0 < H_; k0 += 32) {            // entity-embedding half
    const v16h bm = fragB(wr + k0);
    acc0 = wmma_f16(fragA(hs0 + k0), bm, acc0);
    acc1 = wmma_f16(fragA(hs1 + k0), bm, acc1);
  }
  for (int k0 = H_; k0 < K2H_; k0 += 32) {         // rs half
    const v16h bm = fragB(wr + k0);
    acc0 = wmma_f16(fragA(rs0 + (k0 - H_)), bm, acc0);
    acc1 = wmma_f16(fragA(rs1 + (k0 - H_)), bm, acc1);
  }
#pragma unroll
  for (int r = 0; r < 8; ++r) {
    const int o0 = tm * 32 + hi * 8 + r;
    zout[(long)o0 * H_ + col]        = (_Float16)tanhf(acc0[r] + bias[col]);
    zout[(long)(o0 + 16) * H_ + col] = (_Float16)tanhf(acc1[r] + bias[col]);
  }
}

// ---------------------------------------------------------------------------
// Stage 5: fused bilinear + classifier, bl (434 MB) never materialized:
//   logits[g,n] = bb[n] + sum_{blk,i,j} zh[g,blk*64+i]*zt[g,blk*64+j]*Wb[n,..]
// Key reuse: within a blk, the zt A-vector depends only on (blk, jb), so the
// 4 zt fragments are held in REGISTERS across 64 consecutive WMMAs; zh scalars
// arrive 8-per-v8h register load. Per inner step: 2 global_load_b128 (Wb,
// shared across the two M subtiles) + 16 v_pk_mul_f16 + 2 wmma.
// grid = (NLP/16, 69), block = 32. No LDS needed.
// ---------------------------------------------------------------------------
__global__ void logits_kernel(const _Float16* __restrict__ zh,   // (2208,H)
                              const _Float16* __restrict__ zt,   // (2208,H)
                              const _Float16* __restrict__ Wb16, // (112,KTOT)
                              const float* __restrict__ bb,      // (NL,)
                              float* __restrict__ out) {         // (2208,NL)
  const int tn = blockIdx.x, tm = blockIdx.y;
  const int lane = threadIdx.x;
  const int mrow = lane & 15, hi = lane >> 4;
  const int kbA = hi ? 8 : 0, kbB = hi ? 8 : 0;
  const int col = tn * 16 + mrow;                  // < 112: padded Wb rows
  const int g0 = tm * 32 + mrow, g1 = g0 + 16;

  const _Float16* zh0 = zh + (long)g0 * H_;
  const _Float16* zh1 = zh + (long)g1 * H_;
  const _Float16* zt0 = zt + (long)g0 * H_ + kbA;
  const _Float16* zt1 = zt + (long)g1 * H_ + kbA;
  const _Float16* wr  = Wb16 + (long)col * KTOT + kbB;

  v8f acc0 = {}, acc1 = {};
  for (int blk = 0; blk < NBLK; ++blk) {
    // zt vector fragments: invariant over the 64-wide i sweep of this block
    const v16h t00 = fragA(zt0 + blk * 64);        // subtile 0, j in [0,32)
    const v16h t01 = fragA(zt0 + blk * 64 + 32);   // subtile 0, j in [32,64)
    const v16h t10 = fragA(zt1 + blk * 64);
    const v16h t11 = fragA(zt1 + blk * 64 + 32);
    for (int ig = 0; ig < 8; ++ig) {
      const v8h z0 = *reinterpret_cast<const v8h*>(zh0 + blk * 64 + ig * 8);
      const v8h z1 = *reinterpret_cast<const v8h*>(zh1 + blk * 64 + ig * 8);
#pragma unroll
      for (int ii = 0; ii < 8; ++ii) {
        const int k0 = blk * 4096 + (ig * 8 + ii) * 64;
        const _Float16 a0 = z0[ii], a1 = z1[ii];
        const v16h bmL = fragB(wr + k0);           // j in [0,32)
        acc0 = wmma_f16(t00 * a0, bmL, acc0);
        acc1 = wmma_f16(t10 * a1, bmL, acc1);
        const v16h bmH = fragB(wr + k0 + 32);      // j in [32,64)
        acc0 = wmma_f16(t01 * a0, bmH, acc0);
        acc1 = wmma_f16(t11 * a1, bmH, acc1);
      }
    }
  }
#pragma unroll
  for (int r = 0; r < 8; ++r) {
    const int o0 = tm * 32 + hi * 8 + r;
    if (col < NL_) {
      out[(long)o0 * NL_ + col]        = acc0[r] + bb[col];
      out[(long)(o0 + 16) * NL_ + col] = acc1[r] + bb[col];
    }
  }
}

// ---------------------------------------------------------------------------
// Host launcher
// ---------------------------------------------------------------------------
extern "C" void kernel_launch(void* const* d_in, const int* in_sizes, int n_in,
                              void* d_out, int out_size, void* d_ws, size_t ws_size,
                              hipStream_t stream) {
  (void)in_sizes; (void)n_in; (void)out_size; (void)ws_size;
  const float* seq   = (const float*)d_in[0];
  const float* att   = (const float*)d_in[1];
  const float* mmask = (const float*)d_in[2];
  const float* Wh    = (const float*)d_in[3];
  const float* bh    = (const float*)d_in[4];
  const float* Wt    = (const float*)d_in[5];
  const float* bt    = (const float*)d_in[6];
  const float* Wb    = (const float*)d_in[7];
  const float* bb    = (const float*)d_in[8];
  const int*   midx  = (const int*)d_in[9];
  const int*   hts   = (const int*)d_in[10];
  float* out = (float*)d_out;

  char* ws = (char*)d_ws;
  size_t off = 0;
  auto alloc = [&](size_t bytes) -> void* {
    void* p = ws + off;
    off += (bytes + 255) & ~(size_t)255;
    return p;
  };
  _Float16* eemb16 = (_Float16*)alloc((size_t)B_ * E_ * H_ * 2);        // 0.15 MB
  float*    eatt   = (float*)alloc((size_t)B_ * E_ * NH_ * L_ * 4);     // 4.7 MB
  _Float16* ht16   = (_Float16*)alloc((size_t)B_ * P_ * L_ * 2);        // 4.5 MB
  _Float16* seqT   = (_Float16*)alloc((size_t)B_ * H_ * L_ * 2);        // 6.3 MB
  _Float16* rs16   = (_Float16*)alloc((size_t)B_ * P_ * H_ * 2);        // 3.4 MB
  _Float16* zh     = (_Float16*)alloc((size_t)B_ * P_ * H_ * 2);        // 3.4 MB
  _Float16* zt     = (_Float16*)alloc((size_t)B_ * P_ * H_ * 2);        // 3.4 MB
  _Float16* wh16   = (_Float16*)alloc((size_t)H_ * K2H_ * 2);           // 2.4 MB
  _Float16* wt16   = (_Float16*)alloc((size_t)H_ * K2H_ * 2);           // 2.4 MB
  _Float16* wb16   = (_Float16*)alloc((size_t)NLP_ * KTOT * 2);         // 11.0 MB

  entity_kernel<<<B_ * E_, 256, 0, stream>>>(seq, att, mmask, midx, eemb16, eatt);
  htatt_kernel<<<B_ * P_, 256, 0, stream>>>(eatt, hts, ht16);
  transpose_cvt_kernel<<<dim3(L_ / 32, H_ / 32, B_), dim3(32, 8), 0, stream>>>(seq, seqT);
  cvt_f16_kernel<<<(H_ * K2H_ + 255) / 256, 256, 0, stream>>>(Wh, wh16, H_ * K2H_);
  cvt_f16_kernel<<<(H_ * K2H_ + 255) / 256, 256, 0, stream>>>(Wt, wt16, H_ * K2H_);
  cvt_wb_kernel<<<(int)(((long)NLP_ * KTOT + 255) / 256), 256, 0, stream>>>(Wb, wb16);
  rs_gemm_kernel<<<dim3(H_ / 16, (P_ + 31) / 32, B_), 32, 0, stream>>>(ht16, seqT, rs16);
  z_gemm_kernel<<<dim3(H_ / 16, (B_ * P_) / 32), 32, 0, stream>>>(eemb16, rs16, wh16, bh, hts, 0, zh);
  z_gemm_kernel<<<dim3(H_ / 16, (B_ * P_) / 32), 32, 0, stream>>>(eemb16, rs16, wt16, bt, hts, 1, zt);
  logits_kernel<<<dim3(NLP_ / 16, (B_ * P_) / 32), 32, 0, stream>>>(zh, zt, wb16, bb, out);
}
